// GCHADecoder_1322849927238
// MI455X (gfx1250) — compile-verified, matrix operand order checked by reading, not run on hardware
//
#include <hip/hip_runtime.h>
#include <stdint.h>

// ---------------------------------------------------------------------------
// Problem constants (from reference): B=8, NQ=300, NK=4096, E=256, H=8,
// F=2048, L=6, D=32, scale = 1/sqrt(32)
// ---------------------------------------------------------------------------
#define BB   8
#define NQ_  300
#define NK_  4096
#define EE   256
#define HH   8
#define FF   2048
#define LL   6
#define SCALE_F 0.1767766952966369f

typedef __attribute__((ext_vector_type(16))) __bf16 v16bf;
typedef __attribute__((ext_vector_type(8)))  float  v8f;

union FragU { v16bf b; uint32_t u[8]; uint16_t h[16]; };

__device__ __forceinline__ uint16_t f2bf(float f) {
  uint32_t u = __float_as_uint(f);
  uint32_t r = u + 0x7FFFu + ((u >> 16) & 1u);   // round-to-nearest-even
  return (uint16_t)(r >> 16);
}

__device__ __forceinline__ v8f wmma_bf16(v16bf a, v16bf b, v8f c) {
  return __builtin_amdgcn_wmma_f32_16x16x32_bf16(
      /*neg_a=*/false, a, /*neg_b=*/false, b,
      /*c_mod=*/(short)0, c, /*reuse_a=*/false, /*reuse_b=*/false);
}

// gfx1250 async DMA: global -> LDS, 16 bytes per lane, tracked by ASYNCcnt.
__device__ __forceinline__ void async_copy_b128(uint16_t* lds_dst, const uint16_t* gsrc) {
  uint32_t l = (uint32_t)(uintptr_t)lds_dst;   // low 32 bits of generic = LDS offset
  asm volatile("global_load_async_to_lds_b128 %0, %1, off"
               :: "v"(l), "v"(gsrc)
               : "memory");
}
__device__ __forceinline__ void wait_async0() {
  asm volatile("s_wait_asynccnt 0x0" ::: "memory");
}

// A-fragment (and B-fragment from a [N,K] row-major source): element pairs
// (r, kb),(r, kb+1).  ISA 16-bit A 16x32 layout: lanes 0-15 hold K groups
// {0,1},{2,3},{4,5},{6,7},{16,17}..{22,23}; lanes 16-31 same +8.
__device__ __forceinline__ v16bf load_frag_rowK(const uint16_t* p, int ld, int rbound) {
  const int lane = threadIdx.x & 31;
  const int r  = lane & 15;
  const int hi = lane >> 4;
  FragU f;
#pragma unroll
  for (int v = 0; v < 8; ++v) {
    const int kb = ((v < 4) ? 2 * v : 16 + 2 * (v - 4)) + 8 * hi;
    if (r < rbound) {
      f.u[v] = *(const uint32_t*)(p + (size_t)r * ld + kb);  // 2 bf16, 4B aligned
    } else {
      f.u[v] = 0u;
    }
  }
  return f.b;
}

// B-fragment from a [K,N] row-major source (used for V in P@V): element
// (kb+j, n) with n = lane&15.  Same K packing as A layout.
__device__ __forceinline__ v16bf load_frag_colK(const uint16_t* p, int ld, int kbound) {
  const int lane = threadIdx.x & 31;
  const int n  = lane & 15;
  const int hi = lane >> 4;
  FragU f;
#pragma unroll
  for (int v = 0; v < 8; ++v) {
    const int kb = ((v < 4) ? 2 * v : 16 + 2 * (v - 4)) + 8 * hi;
    f.h[2 * v]     = (kb     < kbound) ? p[(size_t)(kb)     * ld + n] : (uint16_t)0;
    f.h[2 * v + 1] = (kb + 1 < kbound) ? p[(size_t)(kb + 1) * ld + n] : (uint16_t)0;
  }
  return f.b;
}

// ---------------------------------------------------------------------------
// f32 -> bf16 conversion (weights, activations)
// ---------------------------------------------------------------------------
__global__ void cvt_kernel(const float* __restrict__ in, uint16_t* __restrict__ out, int n) {
  int i = blockIdx.x * blockDim.x + threadIdx.x;
  if (i < n) out[i] = f2bf(in[i]);
}

__global__ void init_t_kernel(const float* __restrict__ in, float* __restrict__ outf,
                              uint16_t* __restrict__ outb, int n) {
  int i = blockIdx.x * blockDim.x + threadIdx.x;
  if (i < n) { float v = in[i]; outf[i] = v; outb[i] = f2bf(v); }
}

// ---------------------------------------------------------------------------
// C[M,N] = A[M,K](bf16) @ W[N,K]^T(bf16) + bias,  optional ReLU.
// 256 threads = 8 waves; block tile 64x256; wave tile 32x64; K step 32.
// Double-buffered LDS staging via GLOBAL_LOAD_ASYNC_TO_LDS_B128 so the DMA of
// K-step i+1 overlaps the 8 WMMAs of K-step i.  N must be a multiple of 256
// (true here: 256/768/2048); OOB A rows are clamped (their outputs are never
// stored), keeping EXEC all-ones for the async issues and the WMMAs.
// ---------------------------------------------------------------------------
#define GBM 64
#define GBN 256
__global__ __launch_bounds__(256) void gemm_bias_kernel(
    const uint16_t* __restrict__ A, const uint16_t* __restrict__ W,
    const float* __restrict__ bias,
    float* __restrict__ Cf, uint16_t* __restrict__ Cb,
    int M, int N, int K, int act) {
  __shared__ __align__(16) uint16_t As[2][GBM * 32];   // 2 x 4 KB
  __shared__ __align__(16) uint16_t Ws[2][GBN * 32];   // 2 x 16 KB

  const int tid  = threadIdx.x;
  const int wave = tid >> 5;
  const int wm   = wave >> 2;   // 2 waves along M  -> 64 rows
  const int wn   = wave & 3;    // 4 waves along N  -> 256 cols
  const int bm0  = blockIdx.x * GBM;
  const int bn0  = blockIdx.y * GBN;

  // per-thread staging coordinates (16B = 8 bf16 chunks)
  const int ar = tid >> 2, ac = (tid & 3) * 8;                 // A: 1 chunk/thread
  int agr = bm0 + ar; if (agr >= M) agr = M - 1;               // clamp OOB rows

  auto stage = [&](int buf, int k0) {
    async_copy_b128(&As[buf][ar * 32 + ac], A + (size_t)agr * K + k0 + ac);
#pragma unroll
    for (int i = 0; i < 4; ++i) {                              // W: 4 chunks/thread
      const int ch = i * 256 + tid;
      const int r = ch >> 2, c = (ch & 3) * 8;
      int gr = bn0 + r; if (gr >= N) gr = N - 1;
      async_copy_b128(&Ws[buf][r * 32 + c], W + (size_t)gr * K + k0 + c);
    }
  };

  v8f acc[2][4];
#pragma unroll
  for (int mi = 0; mi < 2; ++mi)
#pragma unroll
    for (int ni = 0; ni < 4; ++ni) acc[mi][ni] = (v8f){};

  stage(0, 0);
  int buf = 0;
  for (int k0 = 0; k0 < K; k0 += 32) {
    wait_async0();        // this wave's DMA for `buf` done
    __syncthreads();      // ...and every other wave's too; prior reads retired
    if (k0 + 32 < K) stage(buf ^ 1, k0 + 32);   // prefetch overlaps compute

    v16bf a0 = load_frag_rowK(&As[buf][(wm * 32 +  0) * 32], 32, 16);
    v16bf a1 = load_frag_rowK(&As[buf][(wm * 32 + 16) * 32], 32, 16);
    v16bf bfr[4];
#pragma unroll
    for (int ni = 0; ni < 4; ++ni)
      bfr[ni] = load_frag_rowK(&Ws[buf][(wn * 64 + ni * 16) * 32], 32, 16);

#pragma unroll
    for (int ni = 0; ni < 4; ++ni) {
      acc[0][ni] = wmma_bf16(a0, bfr[ni], acc[0][ni]);
      acc[1][ni] = wmma_bf16(a1, bfr[ni], acc[1][ni]);
    }
    buf ^= 1;
  }

  const int lane = tid & 31, nl = lane & 15, hi = lane >> 4;
#pragma unroll
  for (int mi = 0; mi < 2; ++mi) {
#pragma unroll
    for (int ni = 0; ni < 4; ++ni) {
#pragma unroll
      for (int v = 0; v < 8; ++v) {
        const int m = bm0 + wm * 32 + mi * 16 + v + 8 * hi;  // C layout: lane hi -> M+8
        const int n = bn0 + wn * 64 + ni * 16 + nl;
        if (m < M && n < N) {
          float val = acc[mi][ni][v];
          if (bias) val += bias[n];
          if (act)  val = fmaxf(val, 0.0f);
          const size_t idx = (size_t)m * N + n;
          if (Cf) Cf[idx] = val;
          if (Cb) Cb[idx] = f2bf(val);
        }
      }
    }
  }
}

// ---------------------------------------------------------------------------
// Flash attention: one wave handles (b, h, 16 query rows).  K/V streamed in
// 32-row chunks; online softmax; score tensor never touches memory.
// Serves both self-attn (nk=300, mask=null) and masked cross-attn (nk=4096).
// ---------------------------------------------------------------------------
__global__ __launch_bounds__(32) void attn_kernel(
    const uint16_t* __restrict__ Qb, int ldq,
    const uint16_t* __restrict__ Kb, int ldk,
    const uint16_t* __restrict__ Vb, int ldv,
    const unsigned char* __restrict__ mask,   // [B,NQ,nk] bool, or null
    float scale, int nq, int nk,
    uint16_t* __restrict__ Out, int ldo) {
  __shared__ __align__(16) uint16_t Pls[16 * 32];

  const int b  = blockIdx.z;
  const int h  = blockIdx.y;
  const int m0 = blockIdx.x * 16;
  const int lane = threadIdx.x & 31, nl = lane & 15, hi = lane >> 4;

  const uint16_t* Q = Qb + (size_t)b * nq * ldq + (size_t)h * 32;
  const uint16_t* Kp = Kb + (size_t)b * nk * ldk + (size_t)h * 32;
  const uint16_t* Vp = Vb + (size_t)b * nk * ldv + (size_t)h * 32;

  const v16bf qf = load_frag_rowK(Q + (size_t)m0 * ldq, ldq, nq - m0);

  float mrow[8], lrow[8];
#pragma unroll
  for (int v = 0; v < 8; ++v) { mrow[v] = -1e30f; lrow[v] = 0.0f; }
  v8f o0 = {}, o1 = {};

  for (int kt = 0; kt < nk; kt += 32) {
    const v16bf k0f = load_frag_rowK(Kp + (size_t)kt * ldk, ldk, nk - kt);
    const v16bf k1f = load_frag_rowK(Kp + (size_t)(kt + 16) * ldk, ldk, nk - kt - 16);

    v8f z = {};
    v8f s0 = wmma_bf16(qf, k0f, z);
    v8f s1 = wmma_bf16(qf, k1f, z);

    const int n0 = kt + nl, n1 = kt + 16 + nl;
    float p0v[8], p1v[8];
#pragma unroll
    for (int v = 0; v < 8; ++v) {
      const int mg = m0 + v + 8 * hi;
      float s0e = s0[v] * scale;
      float s1e = s1[v] * scale;
      bool ok0 = n0 < nk, ok1 = n1 < nk;
      if (mask && mg < nq) {
        const unsigned char* mr = mask + ((size_t)b * nq + mg) * nk;
        if (ok0) ok0 = (mr[n0] != 0);
        if (ok1) ok1 = (mr[n1] != 0);
      }
      if (!ok0) s0e = -1e30f;
      if (!ok1) s1e = -1e30f;

      // row max across the 16 lanes holding this row's columns
      float rm = fmaxf(s0e, s1e);
      rm = fmaxf(rm, __shfl_xor(rm, 1));
      rm = fmaxf(rm, __shfl_xor(rm, 2));
      rm = fmaxf(rm, __shfl_xor(rm, 4));
      rm = fmaxf(rm, __shfl_xor(rm, 8));

      const float mnew  = fmaxf(mrow[v], rm);
      const float alpha = __expf(mrow[v] - mnew);
      const float p0 = __expf(s0e - mnew);
      const float p1 = __expf(s1e - mnew);

      float rs = p0 + p1;
      rs += __shfl_xor(rs, 1);
      rs += __shfl_xor(rs, 2);
      rs += __shfl_xor(rs, 4);
      rs += __shfl_xor(rs, 8);

      lrow[v] = lrow[v] * alpha + rs;
      mrow[v] = mnew;
      o0[v] *= alpha;
      o1[v] *= alpha;
      p0v[v] = p0;
      p1v[v] = p1;
    }

    // C-layout -> A-layout for P via 1KB LDS staging
#pragma unroll
    for (int v = 0; v < 8; ++v) {
      Pls[(v + 8 * hi) * 32 + nl]      = f2bf(p0v[v]);
      Pls[(v + 8 * hi) * 32 + 16 + nl] = f2bf(p1v[v]);
    }
    __syncthreads();
    const v16bf pf  = load_frag_rowK(Pls, 32, 16);
    const v16bf v0f = load_frag_colK(Vp + (size_t)kt * ldv,      ldv, nk - kt); // d 0..15
    const v16bf v1f = load_frag_colK(Vp + (size_t)kt * ldv + 16, ldv, nk - kt); // d 16..31
    o0 = wmma_bf16(pf, v0f, o0);
    o1 = wmma_bf16(pf, v1f, o1);
    __syncthreads();
  }

#pragma unroll
  for (int v = 0; v < 8; ++v) {
    const int m = m0 + v + 8 * hi;
    if (m < nq) {
      const float inv = 1.0f / lrow[v];
      const size_t o = ((size_t)b * nq + m) * ldo + (size_t)h * 32;
      Out[o + nl]      = f2bf(o0[v] * inv);
      Out[o + 16 + nl] = f2bf(o1[v] * inv);
    }
  }
}

// ---------------------------------------------------------------------------
// y = LayerNorm(x + res) * g + b ; writes f32 (residual chain) and bf16.
// One row (E=256) per 256-thread block.
// ---------------------------------------------------------------------------
__global__ __launch_bounds__(256) void ln_kernel(
    const float* __restrict__ x, const float* __restrict__ res,
    const float* __restrict__ g, const float* __restrict__ be,
    float* __restrict__ outf, uint16_t* __restrict__ outb) {
  __shared__ float red[8];
  const int row = blockIdx.x;
  const int t = threadIdx.x;
  const size_t idx = (size_t)row * EE + t;

  float v = x[idx];
  if (res) v += res[idx];

  float s = v;
#pragma unroll
  for (int off = 16; off; off >>= 1) s += __shfl_xor(s, off);
  if ((t & 31) == 0) red[t >> 5] = s;
  __syncthreads();
  float tot = 0.0f;
#pragma unroll
  for (int i = 0; i < 8; ++i) tot += red[i];
  const float mean = tot * (1.0f / EE);
  __syncthreads();

  const float d = v - mean;
  float s2 = d * d;
#pragma unroll
  for (int off = 16; off; off >>= 1) s2 += __shfl_xor(s2, off);
  if ((t & 31) == 0) red[t >> 5] = s2;
  __syncthreads();
  float tv = 0.0f;
#pragma unroll
  for (int i = 0; i < 8; ++i) tv += red[i];
  const float var = tv * (1.0f / EE);

  const float y = d * rsqrtf(var + 1e-5f) * g[t] + be[t];
  outf[idx] = y;
  if (outb) outb[idx] = f2bf(y);
}

// ---------------------------------------------------------------------------
// Host orchestration
// ---------------------------------------------------------------------------
extern "C" void kernel_launch(void* const* d_in, const int* in_sizes, int n_in,
                              void* d_out, int out_size, void* d_ws, size_t ws_size,
                              hipStream_t stream) {
  (void)in_sizes; (void)n_in; (void)out_size; (void)ws_size;

  const float* tgt   = (const float*)d_in[0];
  const float* mem   = (const float*)d_in[1];
  const unsigned char* gmask = (const unsigned char*)d_in[2];  // bool
  const float* sa_wqkv = (const float*)d_in[3];
  const float* sa_bqkv = (const float*)d_in[4];
  const float* sa_wo   = (const float*)d_in[5];
  const float* sa_bo   = (const float*)d_in[6];
  const float* ca_wq   = (const float*)d_in[7];
  const float* ca_bq   = (const float*)d_in[8];
  const float* ca_wk   = (const float*)d_in[9];
  const float* ca_bk   = (const float*)d_in[10];
  const float* ca_wv   = (const float*)d_in[11];
  const float* ca_bv   = (const float*)d_in[12];
  const float* ca_wo   = (const float*)d_in[13];
  const float* ca_bo   = (const float*)d_in[14];
  const float* f_w1    = (const float*)d_in[15];
  const float* f_b1    = (const float*)d_in[16];
  const float* f_w2    = (const float*)d_in[17];
  const float* f_b2    = (const float*)d_in[18];
  const float* ln1g    = (const float*)d_in[19];
  const float* ln1b    = (const float*)d_in[20];
  const float* ln2g    = (const float*)d_in[21];
  const float* ln2b    = (const float*)d_in[22];
  const float* ln3g    = (const float*)d_in[23];
  const float* ln3b    = (const float*)d_in[24];
  const float* lnfg    = (const float*)d_in[25];
  const float* lnfb    = (const float*)d_in[26];

  const int Mq = BB * NQ_;   // 2400 query rows
  const int Mk = BB * NK_;   // 32768 memory rows

  // workspace carve-up (256B aligned slots)
  size_t off = 0;
  auto carve = [&](size_t bytes) { size_t o = off; off = (off + bytes + 255) & ~(size_t)255; return o; };
  char* ws = (char*)d_ws;
  float*    t_f32    = (float*)   (ws + carve((size_t)Mq * EE * 4));
  uint16_t* t_bf16   = (uint16_t*)(ws + carve((size_t)Mq * EE * 2));
  uint16_t* mem_bf16 = (uint16_t*)(ws + carve((size_t)Mk * EE * 2));
  uint16_t* qkv_bf16 = (uint16_t*)(ws + carve((size_t)Mq * 3 * EE * 2));
  uint16_t* att_bf16 = (uint16_t*)(ws + carve((size_t)Mq * EE * 2));
  float*    y_f32    = (float*)   (ws + carve((size_t)Mq * EE * 4));
  uint16_t* q_bf16   = (uint16_t*)(ws + carve((size_t)Mq * EE * 2));
  uint16_t* k_bf16   = (uint16_t*)(ws + carve((size_t)Mk * EE * 2));
  uint16_t* v_bf16   = (uint16_t*)(ws + carve((size_t)Mk * EE * 2));
  uint16_t* ffn_bf16 = (uint16_t*)(ws + carve((size_t)Mq * FF * 2));
  uint16_t* wbuf     = (uint16_t*)(ws + carve((size_t)FF * EE * 2));

  auto gemm = [&](const uint16_t* A, const float* Wf, int wElems, const float* bias,
                  int M, int N, int K, float* Cf, uint16_t* Cb, int act) {
    cvt_kernel<<<(wElems + 255) / 256, 256, 0, stream>>>(Wf, wbuf, wElems);
    dim3 g((M + GBM - 1) / GBM, (N + GBN - 1) / GBN);
    gemm_bias_kernel<<<g, 256, 0, stream>>>(A, wbuf, bias, Cf, Cb, M, N, K, act);
  };

  // init activations + memory once
  init_t_kernel<<<(Mq * EE + 255) / 256, 256, 0, stream>>>(tgt, t_f32, t_bf16, Mq * EE);
  cvt_kernel<<<(Mk * EE + 255) / 256, 256, 0, stream>>>(mem, mem_bf16, Mk * EE);

  const dim3 ga((NQ_ + 15) / 16, HH, BB);

  for (int l = 0; l < LL; ++l) {
    // ---- self attention ----
    gemm(t_bf16, sa_wqkv + (size_t)l * 3 * EE * EE, 3 * EE * EE,
         sa_bqkv + (size_t)l * 3 * EE, Mq, 3 * EE, EE, nullptr, qkv_bf16, 0);
    attn_kernel<<<ga, 32, 0, stream>>>(
        qkv_bf16,            3 * EE,
        qkv_bf16 + EE,       3 * EE,
        qkv_bf16 + 2 * EE,   3 * EE,
        nullptr, SCALE_F, NQ_, NQ_, att_bf16, EE);
    gemm(att_bf16, sa_wo + (size_t)l * EE * EE, EE * EE,
         sa_bo + (size_t)l * EE, Mq, EE, EE, y_f32, nullptr, 0);
    ln_kernel<<<Mq, 256, 0, stream>>>(t_f32, y_f32,
        ln1g + (size_t)l * EE, ln1b + (size_t)l * EE, t_f32, t_bf16);

    // ---- cross attention ----
    gemm(t_bf16,   ca_wq + (size_t)l * EE * EE, EE * EE, ca_bq + (size_t)l * EE,
         Mq, EE, EE, nullptr, q_bf16, 0);
    gemm(mem_bf16, ca_wk + (size_t)l * EE * EE, EE * EE, ca_bk + (size_t)l * EE,
         Mk, EE, EE, nullptr, k_bf16, 0);
    gemm(mem_bf16, ca_wv + (size_t)l * EE * EE, EE * EE, ca_bv + (size_t)l * EE,
         Mk, EE, EE, nullptr, v_bf16, 0);
    attn_kernel<<<ga, 32, 0, stream>>>(
        q_bf16, EE, k_bf16, EE, v_bf16, EE,
        gmask, SCALE_F, NQ_, NK_, att_bf16, EE);
    gemm(att_bf16, ca_wo + (size_t)l * EE * EE, EE * EE,
         ca_bo + (size_t)l * EE, Mq, EE, EE, y_f32, nullptr, 0);
    ln_kernel<<<Mq, 256, 0, stream>>>(t_f32, y_f32,
        ln2g + (size_t)l * EE, ln2b + (size_t)l * EE, t_f32, t_bf16);

    // ---- FFN ----
    gemm(t_bf16, f_w1 + (size_t)l * FF * EE, FF * EE, f_b1 + (size_t)l * FF,
         Mq, FF, EE, nullptr, ffn_bf16, 1 /*relu*/);
    gemm(ffn_bf16, f_w2 + (size_t)l * EE * FF, EE * FF, f_b2 + (size_t)l * EE,
         Mq, EE, FF, y_f32, nullptr, 0);
    ln_kernel<<<Mq, 256, 0, stream>>>(t_f32, y_f32,
        ln3g + (size_t)l * EE, ln3b + (size_t)l * EE, t_f32, t_bf16);
  }

  // final LayerNorm -> d_out (f32)
  ln_kernel<<<Mq, 256, 0, stream>>>(t_f32, nullptr, lnfg, lnfb, (float*)d_out, nullptr);
}
